// TransformerEncoder_62319975465543
// MI455X (gfx1250) — compile-verified, matrix operand order checked by reading, not run on hardware
//
#include <hip/hip_runtime.h>
#include <hip/hip_bf16.h>
#include <math.h>

// ---------------------------------------------------------------------------
// MI455X transformer encoder layer, bf16 WMMA with fp32 accumulation.
// ~58 GFLOP vs ~218 MB fp32 weight traffic -> HBM floor ~9.5us @ 23.3 TB/s;
// bf16 v_wmma_f32_16x16x32_bf16 keeps compute in the same range.
// LDS tiles are K-contiguous (fragments = 2x ds_load_b128), double-buffered.
// bf16->bf16 tile staging in the attention kernels uses the CDNA5 async
// global->LDS engine (ASYNCcnt); fp32->bf16 staging keeps the VGPR path.
// ---------------------------------------------------------------------------

typedef __attribute__((ext_vector_type(16))) __bf16 v16bf;
typedef __attribute__((ext_vector_type(8)))  __bf16 v8bf;
typedef __attribute__((ext_vector_type(8)))  float  v8f;

#define LQ 512
#define DM 2048
#define NH 16
#define DH 128
#define FFD 8192

__device__ __forceinline__ v8f wmma_bf16(const v16bf& a, const v16bf& b, const v8f& c) {
    return __builtin_amdgcn_wmma_f32_16x16x32_bf16(false, a, false, b, (short)0, c, false, false);
}

// ISA 16-bit operand layout: lane half owns K = half*8..+7 (VGPR0-3) and
// K = 16+half*8..+7 (VGPR4-7).  K-contiguous LDS -> two aligned 16B loads.
__device__ __forceinline__ v16bf frag_from_lds(const __bf16* rowp, int half) {
    const v8bf lo = *(const v8bf*)(rowp + half * 8);
    const v8bf hi = *(const v8bf*)(rowp + 16 + half * 8);
    return __builtin_shufflevector(lo, hi, 0, 1, 2, 3, 4, 5, 6, 7,
                                           8, 9, 10, 11, 12, 13, 14, 15);
}

// CDNA5 async global->LDS copy (16B per lane, tracked by ASYNCcnt).
// LDS address = low 32 bits of the generic pointer (flat->LDS aperture
// truncation, ISA 10.2).
__device__ __forceinline__ void async_copy_b128(const void* gptr, void* lptr) {
    const unsigned lds = (unsigned)(size_t)lptr;
    asm volatile("global_load_async_to_lds_b128 %0, %1, off"
                 :: "v"(lds), "v"(gptr) : "memory");
}
__device__ __forceinline__ void wait_async() {
    asm volatile("s_wait_asynccnt 0x0" ::: "memory");
}

__device__ __forceinline__ float gelu_tanh(float x) {
    const float k0 = 0.7978845608028654f;  // sqrt(2/pi)
    const float k1 = 0.044715f;
    float t = tanhf(k0 * (x + k1 * x * x * x));
    return 0.5f * x * (1.0f + t);
}

// ---------------------------------------------------------------------------
// C = act(A[512xK] @ B[KxN] + bias) (+ res).  Block tile 128x128, K-step 32,
// double-buffered LDS.  8 wave32 waves; wave grid 4(M) x 2(N), each wave
// 32x64 = 2x4 accumulators.  aT[m][k]; B transposed in LDS: bTt[n][k].
// ---------------------------------------------------------------------------
template <int N, int K, int ACT>
__global__ __launch_bounds__(256) void gemm_bias_kernel(
    const float* __restrict__ A, const float* __restrict__ B,
    const float* __restrict__ bias, const float* __restrict__ res,
    float* __restrict__ C)
{
    static_assert(K % 64 == 0 && N % 128 == 0, "tile assumptions");
    __shared__ __align__(16) __bf16 aT[2][128][40];
    __shared__ __align__(16) __bf16 bTt[2][128][40];

    const int tid  = threadIdx.x;
    const int m0   = blockIdx.y * 128;
    const int n0   = blockIdx.x * 128;
    const int w    = tid >> 5;
    const int lane = tid & 31;
    const int half = lane >> 4;
    const int r    = lane & 15;
    const int wm   = (w & 3) * 32;
    const int wn   = (w >> 2) * 64;

    const int arow = tid >> 3;          // A staging: 32 rows / pass, 8 float4 per row
    const int acol = (tid & 7) * 4;
    const int sn   = tid & 127;         // B staging: column gather, 2 threads per n
    const int sk0  = (tid >> 7) * 16;

    v8f acc[2][4] = {};

    auto stage = [&](int buf, int kc) {
        float4 va[4];
#pragma unroll
        for (int p = 0; p < 4; ++p)
            va[p] = *(const float4*)(&A[(size_t)(m0 + p * 32 + arow) * K + kc + acol]);
        float tb[16];
#pragma unroll
        for (int i = 0; i < 16; ++i)
            tb[i] = B[(size_t)(kc + sk0 + i) * N + n0 + sn];
#pragma unroll
        for (int p = 0; p < 4; ++p) {
            const int rr = p * 32 + arow;
            aT[buf][rr][acol + 0] = (__bf16)va[p].x;
            aT[buf][rr][acol + 1] = (__bf16)va[p].y;
            aT[buf][rr][acol + 2] = (__bf16)va[p].z;
            aT[buf][rr][acol + 3] = (__bf16)va[p].w;
        }
        v8bf t0, t1;
#pragma unroll
        for (int i = 0; i < 8; ++i) { t0[i] = (__bf16)tb[i]; t1[i] = (__bf16)tb[8 + i]; }
        *(v8bf*)&bTt[buf][sn][sk0 + 0] = t0;
        *(v8bf*)&bTt[buf][sn][sk0 + 8] = t1;
    };

    auto compute = [&](int buf) {
        v16bf afr[2], bfr[4];
#pragma unroll
        for (int mt = 0; mt < 2; ++mt)
            afr[mt] = frag_from_lds(&aT[buf][wm + mt * 16 + r][0], half);
#pragma unroll
        for (int nt = 0; nt < 4; ++nt)
            bfr[nt] = frag_from_lds(&bTt[buf][wn + nt * 16 + r][0], half);
#pragma unroll
        for (int mt = 0; mt < 2; ++mt)
#pragma unroll
            for (int nt = 0; nt < 4; ++nt)
                acc[mt][nt] = wmma_bf16(afr[mt], bfr[nt], acc[mt][nt]);
    };

    stage(0, 0);
    __syncthreads();
#pragma unroll 1
    for (int kc = 0; kc < K; kc += 64) {
        stage(1, kc + 32);                       // K%64==0 -> kc+32 < K always
        if (kc + 64 < K)
            __builtin_prefetch(&B[(size_t)(kc + 64 + sk0) * N + n0 + sn], 0, 1);
        compute(0);
        __syncthreads();
        if (kc + 64 < K) stage(0, kc + 64);
        compute(1);
        __syncthreads();
    }

    // Epilogue: C/D layout -> m = base + half*8 + vgpr, n = base + (lane&15)
#pragma unroll
    for (int mt = 0; mt < 2; ++mt) {
#pragma unroll
        for (int nt = 0; nt < 4; ++nt) {
            const int mb = m0 + wm + mt * 16 + half * 8;
            const int nb = n0 + wn + nt * 16 + r;
            const float bv = bias ? bias[nb] : 0.0f;
#pragma unroll
            for (int g = 0; g < 8; ++g) {
                const int m = mb + g;
                float x = acc[mt][nt][g] + bv;
                if (ACT == 1) x = gelu_tanh(x);
                if (res) x += res[(size_t)m * N + nb];
                C[(size_t)m * N + nb] = x;
            }
        }
    }
}

// ---------------------------------------------------------------------------
// X += positional encoding (computed on the fly, matches reference buffer).
// ---------------------------------------------------------------------------
__global__ __launch_bounds__(256) void add_pos_kernel(float* __restrict__ X)
{
    const int l = blockIdx.x, tid = threadIdx.x;
    float* row = X + (size_t)l * DM;
#pragma unroll
    for (int j = 0; j < 8; ++j) {
        const int c = tid + j * 256;
        const float denom = 10000.0f * expf((float)(c & ~1) * (1.0f / (float)DM));
        const float arg = (float)l / denom;
        row[c] += (c & 1) ? cosf(arg) : sinf(arg);
    }
}

// ---------------------------------------------------------------------------
// [L, D] fp32 -> per-head bf16 [H][L][DH]  (head h at column d*16+h)
// ---------------------------------------------------------------------------
__global__ __launch_bounds__(256) void permute_heads_kernel(
    const float* __restrict__ src, __bf16* __restrict__ dst)
{
    const int l = blockIdx.x, tid = threadIdx.x;
#pragma unroll
    for (int j = 0; j < 8; ++j) {
        const int c = tid + j * 256;
        const int h = c & 15, d = c >> 4;
        dst[((size_t)h * LQ + l) * DH + d] = (__bf16)src[(size_t)l * DM + c];
    }
}

// ---------------------------------------------------------------------------
// S[h][lq][lk] = scale * (Qh[h] @ Kh[h]^T).  K-dim = DH = 128.
// B^T-in-LDS == straight row-major copy of Kh rows; staging uses the async
// global->LDS engine (one B128 per 8 elements, no VGPR round-trip).
// ---------------------------------------------------------------------------
__global__ __launch_bounds__(256) void attn_scores_kernel(
    const __bf16* __restrict__ Qh, const __bf16* __restrict__ Kh,
    float* __restrict__ S, float scale)
{
    __shared__ __align__(16) __bf16 aT[2][128][40];
    __shared__ __align__(16) __bf16 bTt[2][128][40];

    const int tid  = threadIdx.x;
    const int h    = blockIdx.z;
    const int m0   = blockIdx.y * 128;
    const int n0   = blockIdx.x * 128;
    const int w    = tid >> 5, lane = tid & 31;
    const int half = lane >> 4, r = lane & 15;
    const int wm   = (w & 3) * 32, wn = (w >> 2) * 64;

    const __bf16* Q  = Qh + (size_t)h * LQ * DH;
    const __bf16* Kp = Kh + (size_t)h * LQ * DH;

    const int srow = tid >> 1;            // 128 rows, 2 threads per row
    const int sk0  = (tid & 1) * 16;

    v8f acc[2][4] = {};

    auto stage = [&](int buf, int kc) {
        async_copy_b128(&Q [(size_t)(m0 + srow) * DH + kc + sk0], &aT [buf][srow][sk0]);
        async_copy_b128(&Q [(size_t)(m0 + srow) * DH + kc + sk0 + 8], &aT [buf][srow][sk0 + 8]);
        async_copy_b128(&Kp[(size_t)(n0 + srow) * DH + kc + sk0], &bTt[buf][srow][sk0]);
        async_copy_b128(&Kp[(size_t)(n0 + srow) * DH + kc + sk0 + 8], &bTt[buf][srow][sk0 + 8]);
    };
    auto compute = [&](int buf) {
        v16bf afr[2], bfr[4];
#pragma unroll
        for (int mt = 0; mt < 2; ++mt)
            afr[mt] = frag_from_lds(&aT[buf][wm + mt * 16 + r][0], half);
#pragma unroll
        for (int nt = 0; nt < 4; ++nt)
            bfr[nt] = frag_from_lds(&bTt[buf][wn + nt * 16 + r][0], half);
#pragma unroll
        for (int mt = 0; mt < 2; ++mt)
#pragma unroll
            for (int nt = 0; nt < 4; ++nt)
                acc[mt][nt] = wmma_bf16(afr[mt], bfr[nt], acc[mt][nt]);
    };

    stage(0, 0);
    wait_async();
    __syncthreads();
#pragma unroll 1
    for (int kc = 0; kc < DH; kc += 64) {
        stage(1, kc + 32);
        compute(0);
        wait_async();
        __syncthreads();
        if (kc + 64 < DH) stage(0, kc + 64);
        compute(1);
        wait_async();
        __syncthreads();
    }

#pragma unroll
    for (int mt = 0; mt < 2; ++mt)
#pragma unroll
        for (int nt = 0; nt < 4; ++nt) {
            const int mb = m0 + wm + mt * 16 + half * 8;
            const int nb = n0 + wn + nt * 16 + r;
#pragma unroll
            for (int g = 0; g < 8; ++g)
                S[((size_t)h * LQ + (mb + g)) * LQ + nb] = acc[mt][nt][g] * scale;
        }
}

// ---------------------------------------------------------------------------
// Row softmax over keys; emits bf16 probabilities for the P@V WMMA.
// ---------------------------------------------------------------------------
__global__ __launch_bounds__(256) void softmax_kernel(
    const float* __restrict__ S, __bf16* __restrict__ P)
{
    const int lq = blockIdx.x, h = blockIdx.y, tid = threadIdx.x;
    const float* row = S + ((size_t)h * LQ + lq) * LQ;
    __bf16* prow = P + ((size_t)h * LQ + lq) * LQ;
    __shared__ float red[256];

    const float v0 = row[tid], v1 = row[tid + 256];
    red[tid] = fmaxf(v0, v1);
    __syncthreads();
    for (int s = 128; s > 0; s >>= 1) { if (tid < s) red[tid] = fmaxf(red[tid], red[tid + s]); __syncthreads(); }
    const float mx = red[0];
    __syncthreads();

    const float e0 = expf(v0 - mx), e1 = expf(v1 - mx);
    red[tid] = e0 + e1;
    __syncthreads();
    for (int s = 128; s > 0; s >>= 1) { if (tid < s) red[tid] += red[tid + s]; __syncthreads(); }
    const float inv = 1.0f / red[0];

    prow[tid]       = (__bf16)(e0 * inv);
    prow[tid + 256] = (__bf16)(e1 * inv);
}

// ---------------------------------------------------------------------------
// Acat[:, d*16+h] = P[h] @ Vh[h]   (M=512, N=128, K=512)
// P tile staged with async global->LDS; V tile is a transpose gather (VGPR).
// ---------------------------------------------------------------------------
__global__ __launch_bounds__(256) void attn_av_kernel(
    const __bf16* __restrict__ P, const __bf16* __restrict__ Vh,
    float* __restrict__ Acat)
{
    __shared__ __align__(16) __bf16 aT[2][128][40];
    __shared__ __align__(16) __bf16 bTt[2][128][40];

    const int tid  = threadIdx.x;
    const int h    = blockIdx.z;
    const int m0   = blockIdx.y * 128;
    const int w    = tid >> 5, lane = tid & 31;
    const int half = lane >> 4, r = lane & 15;
    const int wm   = (w & 3) * 32, wn = (w >> 2) * 64;

    const __bf16* Pp = P + (size_t)h * LQ * LQ;
    const __bf16* Vp = Vh + (size_t)h * LQ * DH;

    const int srow = tid >> 1;           // P staging: row copy
    const int pk0  = (tid & 1) * 16;
    const int sn   = tid & 127;          // V staging: column gather (transpose)
    const int sk0  = (tid >> 7) * 16;

    v8f acc[2][4] = {};

    auto stage = [&](int buf, int kc) {
        async_copy_b128(&Pp[(size_t)(m0 + srow) * LQ + kc + pk0], &aT[buf][srow][pk0]);
        async_copy_b128(&Pp[(size_t)(m0 + srow) * LQ + kc + pk0 + 8], &aT[buf][srow][pk0 + 8]);
        v8bf t0, t1;
#pragma unroll
        for (int i = 0; i < 8; ++i) t0[i] = Vp[(size_t)(kc + sk0 + i) * DH + sn];
#pragma unroll
        for (int i = 0; i < 8; ++i) t1[i] = Vp[(size_t)(kc + sk0 + 8 + i) * DH + sn];
        *(v8bf*)&bTt[buf][sn][sk0 + 0] = t0;
        *(v8bf*)&bTt[buf][sn][sk0 + 8] = t1;
    };
    auto compute = [&](int buf) {
        v16bf afr[2], bfr[4];
#pragma unroll
        for (int mt = 0; mt < 2; ++mt)
            afr[mt] = frag_from_lds(&aT[buf][wm + mt * 16 + r][0], half);
#pragma unroll
        for (int nt = 0; nt < 4; ++nt)
            bfr[nt] = frag_from_lds(&bTt[buf][wn + nt * 16 + r][0], half);
#pragma unroll
        for (int mt = 0; mt < 2; ++mt)
#pragma unroll
            for (int nt = 0; nt < 4; ++nt)
                acc[mt][nt] = wmma_bf16(afr[mt], bfr[nt], acc[mt][nt]);
    };

    stage(0, 0);
    wait_async();
    __syncthreads();
#pragma unroll 1
    for (int kc = 0; kc < LQ; kc += 64) {
        stage(1, kc + 32);
        compute(0);
        wait_async();
        __syncthreads();
        if (kc + 64 < LQ) stage(0, kc + 64);
        compute(1);
        wait_async();
        __syncthreads();
    }

#pragma unroll
    for (int mt = 0; mt < 2; ++mt)
#pragma unroll
        for (int nt = 0; nt < 4; ++nt) {
            const int mb = m0 + wm + mt * 16 + half * 8;
            const int nb = wn + nt * 16 + r;      // n in [0,128)
#pragma unroll
            for (int g = 0; g < 8; ++g)
                Acat[(size_t)(mb + g) * DM + nb * NH + h] = acc[mt][nt][g];
        }
}

// ---------------------------------------------------------------------------
// Row LayerNorm over D=2048.
// ---------------------------------------------------------------------------
__global__ __launch_bounds__(256) void layernorm_kernel(
    const float* __restrict__ X, const float* __restrict__ g,
    const float* __restrict__ b, float* __restrict__ Y)
{
    const int l = blockIdx.x, tid = threadIdx.x;
    const float* row = X + (size_t)l * DM;
    float* out = Y + (size_t)l * DM;
    __shared__ float red[256];

    float vals[8], s = 0.0f;
#pragma unroll
    for (int j = 0; j < 8; ++j) { vals[j] = row[tid + j * 256]; s += vals[j]; }
    red[tid] = s;
    __syncthreads();
    for (int st = 128; st > 0; st >>= 1) { if (tid < st) red[tid] += red[tid + st]; __syncthreads(); }
    const float mu = red[0] * (1.0f / (float)DM);
    __syncthreads();

    float v = 0.0f;
#pragma unroll
    for (int j = 0; j < 8; ++j) { const float d = vals[j] - mu; v += d * d; }
    red[tid] = v;
    __syncthreads();
    for (int st = 128; st > 0; st >>= 1) { if (tid < st) red[tid] += red[tid + st]; __syncthreads(); }
    const float inv = rsqrtf(red[0] * (1.0f / (float)DM) + 1e-5f);

#pragma unroll
    for (int j = 0; j < 8; ++j) {
        const int c = tid + j * 256;
        out[c] = (vals[j] - mu) * inv * g[c] + b[c];
    }
}

// ---------------------------------------------------------------------------
extern "C" void kernel_launch(void* const* d_in, const int* in_sizes, int n_in,
                              void* d_out, int out_size, void* d_ws, size_t ws_size,
                              hipStream_t stream)
{
    (void)in_sizes; (void)n_in; (void)out_size; (void)ws_size;

    const float* seq  = (const float*)d_in[0];
    const float* embW = (const float*)d_in[1];
    const float* embB = (const float*)d_in[2];
    const float* Wq   = (const float*)d_in[3];
    const float* bq   = (const float*)d_in[4];
    const float* Wk   = (const float*)d_in[5];
    const float* bk   = (const float*)d_in[6];
    const float* Wv   = (const float*)d_in[7];
    const float* bv   = (const float*)d_in[8];
    const float* Wo   = (const float*)d_in[9];
    const float* bo   = (const float*)d_in[10];
    const float* lng  = (const float*)d_in[11];
    const float* lnb  = (const float*)d_in[12];
    const float* W1   = (const float*)d_in[13];
    const float* b1   = (const float*)d_in[14];
    const float* W2   = (const float*)d_in[15];
    const float* b2   = (const float*)d_in[16];

    char* ws = (char*)d_ws;
    size_t off = 0;
    auto take = [&](size_t bytes) { char* p = ws + off; off = (off + bytes + 255) & ~(size_t)255; return p; };

    const size_t fLD = (size_t)LQ * DM * sizeof(float);       // 4 MB
    const size_t hLD = (size_t)NH * LQ * DH * sizeof(__bf16); // 2 MB
    const size_t fSS = (size_t)NH * LQ * LQ * sizeof(float);  // 16 MB
    const size_t hSS = (size_t)NH * LQ * LQ * sizeof(__bf16); // 8 MB

    float*  X    = (float*)take(fLD);
    float*  Qf   = (float*)take(fLD);
    float*  Kf   = (float*)take(fLD);
    float*  Vf   = (float*)take(fLD);
    __bf16* Qh   = (__bf16*)take(hLD);
    __bf16* Kh   = (__bf16*)take(hLD);
    __bf16* Vh   = (__bf16*)take(hLD);
    float*  S    = (float*)take(fSS);        // reused as FFN hidden (16 MB)
    __bf16* P    = (__bf16*)take(hSS);
    float*  Xa   = (float*)take(fLD);
    float*  Acat = Qf;                       // Qf/Kf/Vf free after permute
    float*  T0   = Kf;
    float*  Hf   = S;                        // S free after softmax
    float*  T1   = Vf;
    float*  Yout = (float*)d_out;

    const dim3 blk(256);
    const dim3 gD (DM / 128, LQ / 128);      // N=2048 GEMMs
    const dim3 gFF(FFD / 128, LQ / 128);     // N=8192 GEMM

    // 1) embedding
    gemm_bias_kernel<DM, DM, 0><<<gD, blk, 0, stream>>>(seq, embW, embB, nullptr, X);
    // 2) + positional encoding
    add_pos_kernel<<<LQ, blk, 0, stream>>>(X);
    // 3) Q/K/V projections
    gemm_bias_kernel<DM, DM, 0><<<gD, blk, 0, stream>>>(X, Wq, bq, nullptr, Qf);
    gemm_bias_kernel<DM, DM, 0><<<gD, blk, 0, stream>>>(X, Wk, bk, nullptr, Kf);
    gemm_bias_kernel<DM, DM, 0><<<gD, blk, 0, stream>>>(X, Wv, bv, nullptr, Vf);
    // 4) per-head bf16 layout
    permute_heads_kernel<<<LQ, blk, 0, stream>>>(Qf, Qh);
    permute_heads_kernel<<<LQ, blk, 0, stream>>>(Kf, Kh);
    permute_heads_kernel<<<LQ, blk, 0, stream>>>(Vf, Vh);
    // 5) scores = Qh @ Kh^T / sqrt(D)
    attn_scores_kernel<<<dim3(LQ / 128, LQ / 128, NH), blk, 0, stream>>>(
        Qh, Kh, S, 0.022097086912079608f /* 1/sqrt(2048) */);
    // 6) softmax over keys -> bf16 probs
    softmax_kernel<<<dim3(LQ, NH), blk, 0, stream>>>(S, P);
    // 7) attn @ V, scatter back to [L, D] interleaved layout
    attn_av_kernel<<<dim3(1, LQ / 128, NH), blk, 0, stream>>>(P, Vh, Acat);
    // 8) output projection + residual X
    gemm_bias_kernel<DM, DM, 0><<<gD, blk, 0, stream>>>(Acat, Wo, bo, X, T0);
    // 9) LayerNorm -> Xa
    layernorm_kernel<<<LQ, blk, 0, stream>>>(T0, lng, lnb, Xa);
    // 10) FFN1 + GELU
    gemm_bias_kernel<FFD, DM, 1><<<gFF, blk, 0, stream>>>(Xa, W1, b1, nullptr, Hf);
    // 11) FFN2 + residual Xa
    gemm_bias_kernel<DM, FFD, 0><<<gD, blk, 0, stream>>>(Hf, W2, b2, Xa, T1);
    // 12) final LayerNorm -> output
    layernorm_kernel<<<LQ, blk, 0, stream>>>(T1, lng, lnb, Yout);
}
